// DisentangledSelfAttention_28518582846211
// MI455X (gfx1250) — matrix operand order repaired and checked by reading, not compiled, not run on hardware
//
#include <hip/hip_runtime.h>
#include <hip/hip_bf16.h>

// ---------------------------------------------------------------------------
// DeBERTa disentangled self-attention for MI455X (gfx1250), wave32, WMMA f16.
// ---------------------------------------------------------------------------

typedef _Float16 half_t;
typedef __attribute__((ext_vector_type(16))) _Float16 v16h;
typedef __attribute__((ext_vector_type(8)))  float    v8f;

#define NB   2
#define NSEQ 2048
#define NH   16
#define ND   64
#define HIDW 1024
#define SCALE_F (0.07216878364870323f)  // 1/sqrt(3*64)

__device__ __forceinline__ v8f wmma_f16(v16h a, v16h b, v8f c) {
  return __builtin_amdgcn_wmma_f32_16x16x32_f16(false, a, false, b,
                                                (short)0, c, false, false);
}

__device__ __forceinline__ v8f zero8() {
  v8f z = {0.f, 0.f, 0.f, 0.f, 0.f, 0.f, 0.f, 0.f};
  return z;
}

// A-fragment (16x32 f16). base = element (row0, k0) of row-major [ld] tile.
// Lanes 0-15 row=lane: regs hold K 0..7 and 16..23; lanes 16-31: K 8..15, 24..31.
__device__ __forceinline__ v16h load_a_frag(const half_t* base, int ld, int lane) {
  int hl = lane >> 4, r = lane & 15;
  const half_t* pr = base + r * ld;
  v16h out;
  ((uint4*)&out)[0] = *(const uint4*)(pr + hl * 8);
  ((uint4*)&out)[1] = *(const uint4*)(pr + 16 + hl * 8);
  return out;
}

// B-fragment (32x16 f16) read from K-transposed storage: baseT holds element
// (k0, n0) of a [n][k] layout with row stride ld. Lane half h, col c=lane&15:
// regs 0..7 hold B[k = h*8 + i][n = c], regs 8..15 hold B[k = 16 + h*8 + (i-8)][n = c].
__device__ __forceinline__ v16h load_b_fragT(const half_t* baseT, int ld, int lane) {
  int hl = lane >> 4, c = lane & 15;
  const half_t* pc = baseT + c * ld;
  v16h out;
  ((uint4*)&out)[0] = *(const uint4*)(pc + hl * 8);
  ((uint4*)&out)[1] = *(const uint4*)(pc + 16 + hl * 8);
  return out;
}

// Async global->LDS 16-byte copy (GLOBAL_LOAD_ASYNC_TO_LDS_B128, ASYNCcnt).
// ldsoff = LDS byte offset (low 32 bits of generic shared address), g = global.
__device__ __forceinline__ void async_copy_b128(unsigned ldsoff, const half_t* g) {
  asm volatile("global_load_async_to_lds_b128 %0, %1, off"
               :: "v"(ldsoff), "v"(g) : "memory");
}
__device__ __forceinline__ void async_wait0() {
  asm volatile("s_wait_asynccnt 0" ::: "memory");
}

// ---------------------------------------------------------------------------
// fp32 -> f16 convert
// ---------------------------------------------------------------------------
__global__ __launch_bounds__(256) void cvt_f32_to_f16(const float* __restrict__ in,
                                                      half_t* __restrict__ out, int n) {
  for (int i = blockIdx.x * 256 + threadIdx.x; i < n; i += gridDim.x * 256)
    out[i] = (half_t)in[i];
}

// ---------------------------------------------------------------------------
// WMMA GEMM: C[M,N] = A[M,K] * W[K,N] + bias (+ residual).
// Block: 256 threads (8 waves), tile 128x128, K-step 32.
// F32OUT=false: f16 output Ch.  F32OUT=true: f32 output Cf += residual.
// A tile staged via async global->LDS DMA; B tile transposed through VGPRs.
// ---------------------------------------------------------------------------
template <bool F32OUT>
__global__ __launch_bounds__(256) void gemm_wmma_kernel(
    const half_t* __restrict__ A, const half_t* __restrict__ W,
    const float* __restrict__ bias, const float* __restrict__ residual,
    half_t* __restrict__ Ch, float* __restrict__ Cf, int M, int K, int N) {
  __shared__ __align__(16) half_t As[128 * 32];   // [m][k]
  __shared__ __align__(16) half_t Bs[128 * 32];   // transposed: [n][k]

  const int tid  = threadIdx.x;
  const int lane = tid & 31;
  const int wave = tid >> 5;
  const int mblk = blockIdx.y * 128;
  const int nblk = blockIdx.x * 128;
  const int mw = (wave & 3) * 32;   // 2 m-tiles per wave
  const int nw = (wave >> 2) * 64;  // 4 n-tiles per wave

  // Loop-invariant staging coordinates.
  int a_row[2], a_col[2];
  unsigned as_off[2];
  int b_k[2], b_n[2];
  for (int i = 0; i < 2; ++i) {
    int idx = tid * 2 + i;                 // 0..511 chunks of 8 halves
    a_row[i] = idx >> 2; a_col[i] = (idx & 3) * 8;
    as_off[i] = (unsigned)(size_t)&As[a_row[i] * 32 + a_col[i]];
    b_k[i] = idx >> 4;  b_n[i] = (idx & 15) * 8;
  }

  v8f acc[2][4];
  for (int mi = 0; mi < 2; ++mi)
    for (int ni = 0; ni < 4; ++ni) acc[mi][ni] = zero8();

  for (int kb = 0; kb < K; kb += 32) {
    // A tile: async global->LDS DMA (no VGPR round-trip).
    for (int i = 0; i < 2; ++i)
      async_copy_b128(as_off[i], &A[(size_t)(mblk + a_row[i]) * K + kb + a_col[i]]);
    // B tile: both b128 loads issued before any LDS store.
    uint4 w0 = *(const uint4*)&W[(size_t)(kb + b_k[0]) * N + nblk + b_n[0]];
    uint4 w1 = *(const uint4*)&W[(size_t)(kb + b_k[1]) * N + nblk + b_n[1]];
    // Prefetch next K-slice toward near caches (global_prefetch_b8).
    if (kb + 32 < K) {
      __builtin_prefetch(&A[(size_t)(mblk + (tid >> 1)) * K + kb + 32 + (tid & 1) * 16], 0, 3);
      __builtin_prefetch(&W[(size_t)(kb + 32 + (tid >> 3)) * N + nblk + (tid & 7) * 16], 0, 3);
    }
    {
      const half_t* wh0 = (const half_t*)&w0;
      const half_t* wh1 = (const half_t*)&w1;
      for (int e = 0; e < 8; ++e) Bs[(b_n[0] + e) * 32 + b_k[0]] = wh0[e];
      for (int e = 0; e < 8; ++e) Bs[(b_n[1] + e) * 32 + b_k[1]] = wh1[e];
    }
    async_wait0();
    __syncthreads();

    v16h bfr[4];
    for (int ni = 0; ni < 4; ++ni)
      bfr[ni] = load_b_fragT(Bs + (nw + ni * 16) * 32, 32, lane);
    for (int mi = 0; mi < 2; ++mi) {
      v16h a = load_a_frag(As + (mw + mi * 16) * 32, 32, lane);
      for (int ni = 0; ni < 4; ++ni)
        acc[mi][ni] = wmma_f16(a, bfr[ni], acc[mi][ni]);
    }
    __syncthreads();
  }

  // Epilogue. C layout: reg r -> row r + 8*(lane>>4), col = lane&15.
  const int hl = lane >> 4, nn = lane & 15;
  for (int mi = 0; mi < 2; ++mi)
    for (int ni = 0; ni < 4; ++ni) {
      int col = nblk + nw + ni * 16 + nn;
      float bv = bias[col];
      for (int r = 0; r < 8; ++r) {
        int row = mblk + mw + mi * 16 + r + 8 * hl;
        float v = acc[mi][ni][r] + bv;
        if (F32OUT) {
          Cf[(size_t)row * N + col] = v + residual[(size_t)row * N + col];
        } else {
          Ch[(size_t)row * N + col] = (half_t)v;
        }
      }
    }
}

// ---------------------------------------------------------------------------
// Build extended Q/K:  Qx[b,h,n,0:64]=(q+rel_q)*SCALE, [64:128]=q*SCALE
//                      Kx[b,h,n,0:64]=k,               [64:128]=rel_k
// q/k layout from GEMM: [b, n, h*64+d]; rel layout: [n, h*64+d].
// ---------------------------------------------------------------------------
__global__ __launch_bounds__(256) void build_qx_kx_kernel(
    const half_t* __restrict__ q, const half_t* __restrict__ k,
    const half_t* __restrict__ rq, const half_t* __restrict__ rk,
    half_t* __restrict__ Qx, half_t* __restrict__ Kx) {
  size_t idx = (size_t)blockIdx.x * 256 + threadIdx.x;   // over B*H*N*128
  int dp = idx & 127;
  size_t t1 = idx >> 7;
  int n = (int)(t1 % NSEQ);
  size_t t2 = t1 / NSEQ;
  int h = (int)(t2 % NH);
  int b = (int)(t2 / NH);
  int d = dp & 63;
  size_t qk_off  = ((size_t)b * NSEQ + n) * HIDW + h * ND + d;
  size_t rel_off = (size_t)n * HIDW + h * ND + d;
  float qv, kv;
  if (dp < 64) {
    qv = ((float)q[qk_off] + (float)rq[rel_off]) * SCALE_F;
    kv = (float)k[qk_off];
  } else {
    qv = (float)q[qk_off] * SCALE_F;
    kv = (float)rk[rel_off];
  }
  Qx[idx] = (half_t)qv;
  Kx[idx] = (half_t)kv;
}

// ---------------------------------------------------------------------------
// Flash attention: per (b,h), rows in blocks of 64, KV in blocks of 64.
// S = Qx(64x128) * Kx^T  -> online softmax -> O += P * V (64x64).
// Q fragments live in registers across the whole KV loop.
// K tiles are staged with GLOBAL_LOAD_ASYNC_TO_LDS_B128 (ASYNCcnt).
// V tiles: coalesced b128 loads + transposing ds_store_b16.
// ---------------------------------------------------------------------------
__global__ __launch_bounds__(256) void flash_attn_kernel(
    const half_t* __restrict__ Qx,  // [B*H, N, 128]
    const half_t* __restrict__ Kx,  // [B*H, N, 128]
    const half_t* __restrict__ Vg,  // [B, N, H*64]
    half_t* __restrict__ Oh) {      // [B, N, H*64]
  __shared__ __align__(16) half_t Ks[64 * 128];   // [j][k]  (already B^T layout)
  __shared__ __align__(16) half_t Vt[64 * 64];    // transposed: [d][j]
  __shared__ __align__(16) float  Ss[64 * 64];    // scores f32
  __shared__ __align__(16) half_t Ps[64 * 64];    // probabilities f16
  __shared__ float m_run[64], l_run[64], alpha_s[64], mnew_s[64];
  __shared__ float part_m[64 * 4], part_l[64 * 4];

  const int tid  = threadIdx.x;
  const int lane = tid & 31;
  const int wave = tid >> 5;
  const int hl   = lane >> 4;
  const int nn   = lane & 15;

  const int bh = blockIdx.y;          // b*NH + h
  const int b  = bh / NH;
  const int h  = bh % NH;
  const int i0 = blockIdx.x * 64;

  const half_t* Qbh = Qx + (size_t)bh * NSEQ * 128;
  const half_t* Kbh = Kx + (size_t)bh * NSEQ * 128;
  const half_t* Vb  = Vg + (size_t)b * NSEQ * HIDW + h * ND;

  const int mi  = wave >> 1;          // row tile 0..3
  const int nj0 = (wave & 1) * 2;     // col tiles nj0, nj0+1

  // Q fragments for this wave's row tile: 4 K-steps of 32
  v16h qa[4];
  for (int kk = 0; kk < 4; ++kk)
    qa[kk] = load_a_frag(Qbh + (size_t)(i0 + mi * 16) * 128 + kk * 32, 128, lane);

  v8f o[2] = {zero8(), zero8()};

  if (tid < 64) { m_run[tid] = -3.0e38f; l_run[tid] = 0.f; }
  __syncthreads();

  // Loop-invariant staging coordinates for K (async) and V (transpose) tiles.
  unsigned ks_off[2];
  int ks_row[2], ks_col[2];
  int vj[2], vd[2];
  for (int i = 0; i < 2; ++i) {
    int idx = tid * 2 + i;            // 0..511 chunks of 8 halves
    ks_row[i] = idx >> 3;
    ks_col[i] = (idx & 7) * 8;
    ks_off[i] = (unsigned)(size_t)&Ks[ks_row[i] * 128 + ks_col[i]];
    vj[i] = idx >> 3;                 // V tile row (j), 8 chunks per row
    vd[i] = (idx & 7) * 8;            // V tile col start (d)
  }

  for (int j0 = 0; j0 < NSEQ; j0 += 64) {
    // --- K tile via async global->LDS DMA ---
    for (int i = 0; i < 2; ++i)
      async_copy_b128(ks_off[i], &Kbh[(size_t)(j0 + ks_row[i]) * 128 + ks_col[i]]);
    // --- V tile: both b128 loads issued, then transposing LDS stores ---
    uint4 vv0 = *(const uint4*)&Vb[(size_t)(j0 + vj[0]) * HIDW + vd[0]];
    uint4 vv1 = *(const uint4*)&Vb[(size_t)(j0 + vj[1]) * HIDW + vd[1]];
    {
      const half_t* vh0 = (const half_t*)&vv0;
      const half_t* vh1 = (const half_t*)&vv1;
      for (int e = 0; e < 8; ++e) Vt[(vd[0] + e) * 64 + vj[0]] = vh0[e];
      for (int e = 0; e < 8; ++e) Vt[(vd[1] + e) * 64 + vj[1]] = vh1[e];
    }
    async_wait0();
    __syncthreads();

    // --- S = Qx * Kx^T (this wave: tiles (mi, nj0), (mi, nj0+1)) ---
    for (int t = 0; t < 2; ++t) {
      int nj = nj0 + t;
      v8f s = zero8();
      for (int kk = 0; kk < 4; ++kk) {
        v16h bf = load_b_fragT(Ks + (nj * 16) * 128 + kk * 32, 128, lane);
        s = wmma_f16(qa[kk], bf, s);
      }
      for (int r = 0; r < 8; ++r)
        Ss[(mi * 16 + r + 8 * hl) * 64 + nj * 16 + nn] = s[r];
    }
    __syncthreads();

    // --- row stats: 4 threads per row ---
    {
      int r = tid >> 2, q4 = tid & 3;
      float mx = -3.0e38f;
      for (int c = 0; c < 16; ++c) mx = fmaxf(mx, Ss[r * 64 + q4 * 16 + c]);
      part_m[r * 4 + q4] = mx;
    }
    __syncthreads();
    if ((tid & 3) == 0) {
      int r = tid >> 2;
      float m4 = fmaxf(fmaxf(part_m[r * 4], part_m[r * 4 + 1]),
                       fmaxf(part_m[r * 4 + 2], part_m[r * 4 + 3]));
      float mn = fmaxf(m_run[r], m4);
      mnew_s[r] = mn;
      alpha_s[r] = __expf(m_run[r] - mn);
      m_run[r] = mn;
    }
    __syncthreads();

    // --- P = exp(S - m_new), partial row sums ---
    {
      int r = tid >> 2, q4 = tid & 3;
      float mn = mnew_s[r], sum = 0.f;
      for (int c = 0; c < 16; ++c) {
        float p = __expf(Ss[r * 64 + q4 * 16 + c] - mn);
        sum += p;
        Ps[r * 64 + q4 * 16 + c] = (half_t)p;
      }
      part_l[r * 4 + q4] = sum;
    }
    __syncthreads();
    if ((tid & 3) == 0) {
      int r = tid >> 2;
      float s4 = part_l[r * 4] + part_l[r * 4 + 1] + part_l[r * 4 + 2] + part_l[r * 4 + 3];
      l_run[r] = l_run[r] * alpha_s[r] + s4;
    }

    // --- O = O*alpha + P*V (this wave: O tiles (mi, nj0), (mi, nj0+1)) ---
    v16h pa[2];
    for (int kk = 0; kk < 2; ++kk)
      pa[kk] = load_a_frag(Ps + (mi * 16) * 64 + kk * 32, 64, lane);
    for (int t = 0; t < 2; ++t) {
      int dj = nj0 + t;
      for (int r = 0; r < 8; ++r)
        o[t][r] *= alpha_s[mi * 16 + r + 8 * hl];
      for (int kk = 0; kk < 2; ++kk) {
        v16h bf = load_b_fragT(Vt + (dj * 16) * 64 + kk * 32, 64, lane);
        o[t] = wmma_f16(pa[kk], bf, o[t]);
      }
    }
    __syncthreads();   // protect Ks/Vt/Ss/Ps/stats before next iteration
  }

  // --- normalize and store O -> [b, i, h*64+d] (f16) ---
  for (int t = 0; t < 2; ++t) {
    int dj = nj0 + t;
    for (int r = 0; r < 8; ++r) {
      int rl = mi * 16 + r + 8 * hl;
      float ov = o[t][r] / l_run[rl];
      Oh[((size_t)b * NSEQ + i0 + rl) * HIDW + h * ND + dj * 16 + nn] = (half_t)ov;
    }
  }
}

// ---------------------------------------------------------------------------
// LayerNorm over last dim (1024), one block per row.
// ---------------------------------------------------------------------------
__global__ __launch_bounds__(256) void layernorm_kernel(
    const float* __restrict__ X, const float* __restrict__ g,
    const float* __restrict__ beta, float* __restrict__ out) {
  const int row = blockIdx.x;
  const float* xr = X + (size_t)row * HIDW;
  float s = 0.f, s2 = 0.f;
  for (int c = threadIdx.x; c < HIDW; c += 256) {
    float v = xr[c];
    s += v; s2 += v * v;
  }
  for (int off = 16; off; off >>= 1) {
    s  += __shfl_down(s, off, 32);
    s2 += __shfl_down(s2, off, 32);
  }
  __shared__ float rs[8], rs2[8];
  if ((threadIdx.x & 31) == 0) { rs[threadIdx.x >> 5] = s; rs2[threadIdx.x >> 5] = s2; }
  __syncthreads();
  __shared__ float mu_s, inv_s;
  if (threadIdx.x == 0) {
    float a = 0.f, b2 = 0.f;
    for (int i = 0; i < 8; ++i) { a += rs[i]; b2 += rs2[i]; }
    float mu = a / HIDW;
    float var = b2 / HIDW - mu * mu;
    mu_s = mu;
    inv_s = rsqrtf(var + 1e-7f);
  }
  __syncthreads();
  float mu = mu_s, inv = inv_s;
  for (int c = threadIdx.x; c < HIDW; c += 256)
    out[(size_t)row * HIDW + c] = (xr[c] - mu) * inv * g[c] + beta[c];
}

// ---------------------------------------------------------------------------
// Host orchestration
// ---------------------------------------------------------------------------
extern "C" void kernel_launch(void* const* d_in, const int* in_sizes, int n_in,
                              void* d_out, int out_size, void* d_ws, size_t ws_size,
                              hipStream_t stream) {
  const float* x    = (const float*)d_in[0];
  const float* Wq   = (const float*)d_in[1];
  const float* bq   = (const float*)d_in[2];
  const float* Wk   = (const float*)d_in[3];
  const float* bk   = (const float*)d_in[4];
  const float* Wv   = (const float*)d_in[5];
  const float* bv   = (const float*)d_in[6];
  const float* pos  = (const float*)d_in[7];
  const float* Wpq  = (const float*)d_in[8];
  const float* bpq  = (const float*)d_in[9];
  const float* Wpk  = (const float*)d_in[10];
  const float* bpk  = (const float*)d_in[11];
  const float* Wo   = (const float*)d_in[12];
  const float* bo   = (const float*)d_in[13];
  const float* ln_g = (const float*)d_in[14];
  const float* ln_b = (const float*)d_in[15];
  float* out = (float*)d_out;

  const size_t M  = (size_t)NB * NSEQ;        // 4096
  const size_t MH = (size_t)NSEQ;             // 2048 (pos rows)
  const size_t E  = (size_t)HIDW * HIDW;      // weight elems
  const size_t XE = M * HIDW;                 // 4M
  const size_t PE = MH * HIDW;                // 2M
  const size_t QXE = (size_t)NB * NH * NSEQ * 128;  // 8M

  char* p = (char*)d_ws;
  auto carve = [&](size_t bytes) {
    void* r = (void*)p;
    p += (bytes + 255) & ~(size_t)255;
    return r;
  };
  half_t* x_h   = (half_t*)carve(XE * 2);
  half_t* pos_h = (half_t*)carve(PE * 2);
  half_t* wq_h  = (half_t*)carve(E * 2);
  half_t* wk_h  = (half_t*)carve(E * 2);
  half_t* wv_h  = (half_t*)carve(E * 2);
  half_t* wpq_h = (half_t*)carve(E * 2);
  half_t* wpk_h = (half_t*)carve(E * 2);
  half_t* wo_h  = (half_t*)carve(E * 2);
  half_t* q_h   = (half_t*)carve(XE * 2);
  half_t* k_h   = (half_t*)carve(XE * 2);
  half_t* v_h   = (half_t*)carve(XE * 2);
  half_t* rq_h  = (half_t*)carve(PE * 2);
  half_t* rk_h  = (half_t*)carve(PE * 2);
  half_t* qx    = (half_t*)carve(QXE * 2);
  half_t* kx    = (half_t*)carve(QXE * 2);
  half_t* ao_h  = (half_t*)carve(XE * 2);
  float*  ff    = (float*)carve(XE * 4);
  (void)ws_size; (void)n_in; (void)in_sizes; (void)out_size;

  auto cvt = [&](const float* src, half_t* dst, size_t n) {
    int blocks = (int)((n + 255) / 256);
    if (blocks > 16384) blocks = 16384;
    cvt_f32_to_f16<<<blocks, 256, 0, stream>>>(src, dst, (int)n);
  };
  cvt(x, x_h, XE);
  cvt(pos, pos_h, PE);
  cvt(Wq, wq_h, E);  cvt(Wk, wk_h, E);  cvt(Wv, wv_h, E);
  cvt(Wpq, wpq_h, E); cvt(Wpk, wpk_h, E); cvt(Wo, wo_h, E);

  dim3 gX(HIDW / 128, (unsigned)(M / 128));   // (8, 32)
  dim3 gP(HIDW / 128, (unsigned)(MH / 128));  // (8, 16)
  gemm_wmma_kernel<false><<<gX, 256, 0, stream>>>(x_h, wq_h, bq, nullptr, q_h, nullptr,
                                                  (int)M, HIDW, HIDW);
  gemm_wmma_kernel<false><<<gX, 256, 0, stream>>>(x_h, wk_h, bk, nullptr, k_h, nullptr,
                                                  (int)M, HIDW, HIDW);
  gemm_wmma_kernel<false><<<gX, 256, 0, stream>>>(x_h, wv_h, bv, nullptr, v_h, nullptr,
                                                  (int)M, HIDW, HIDW);
  gemm_wmma_kernel<false><<<gP, 256, 0, stream>>>(pos_h, wpq_h, bpq, nullptr, rq_h, nullptr,
                                                  (int)MH, HIDW, HIDW);
  gemm_wmma_kernel<false><<<gP, 256, 0, stream>>>(pos_h, wpk_h, bpk, nullptr, rk_h, nullptr,
                                                  (int)MH, HIDW, HIDW);

  build_qx_kx_kernel<<<(unsigned)(QXE / 256), 256, 0, stream>>>(q_h, k_h, rq_h, rk_h,
                                                                qx, kx);

  dim3 gF(NSEQ / 64, NB * NH);   // (32, 32)
  flash_attn_kernel<<<gF, 256, 0, stream>>>(qx, kx, v_h, ao_h);

  gemm_wmma_kernel<true><<<gX, 256, 0, stream>>>(ao_h, wo_h, bo, x, nullptr, ff,
                                                 (int)M, HIDW, HIDW);

  layernorm_kernel<<<(unsigned)M, 256, 0, stream>>>(ff, ln_g, ln_b, out);
}
